// SparseAxialCausalAttention_11209864643042
// MI455X (gfx1250) — matrix operand order repaired and checked by reading, not compiled, not run on hardware
//
#include <hip/hip_runtime.h>

typedef __bf16 bf16;
typedef __attribute__((ext_vector_type(16))) __bf16 v16bf;
typedef __attribute__((ext_vector_type(8)))  __bf16 v8bf;
typedef __attribute__((ext_vector_type(8)))  float  v8f;

#define NBATCH   16
#define SEQ      1280
#define DIM      1024
#define NHEADS   16
#define DHEAD    64
#define TEXTLEN  256
#define IMGSIDE  32
#define BH       (NBATCH*NHEADS)     /* 256 */
#define NQKV     (3*DIM)             /* 3072 */
#define MROWS    (NBATCH*SEQ)        /* 20480 */
#define QSCALE   0.125f              /* 64^-0.5 */
#define NEGMAX   (-3.4028234663852886e38f)

__device__ __forceinline__ bf16 f2bf(float f) {
  union { float f; unsigned u; } x; x.f = f;
  unsigned r = x.u + 0x7fffu + ((x.u >> 16) & 1u);   // round-to-nearest-even
  unsigned short h = (unsigned short)(r >> 16);
  return __builtin_bit_cast(bf16, h);
}

__device__ __forceinline__ v8f wmma_bf16(v16bf a, v16bf b, v8f c) {
  // D(16x16,f32) = A(16x32,bf16) * B(32x16,bf16) + C
  return __builtin_amdgcn_wmma_f32_16x16x32_bf16(false, a, false, b, (short)0, c,
                                                 false, false);
}

// A fragment 16x32 (MxK), bf16, row-major source (row stride ld elements).
// Layout: lanes 0-15 hold K=0..15, lanes 16-31 hold K=16..31 (matches load_b).
__device__ __forceinline__ v16bf load_a(const bf16* __restrict__ src, int ld, int lane) {
  int m = lane & 15, half = lane >> 4;
  const bf16* r = src + (size_t)m * ld;
  v8bf c0 = *(const v8bf*)(r + 16 * half);
  v8bf c1 = *(const v8bf*)(r + 16 * half + 8);
  v16bf a;
  #pragma unroll
  for (int i = 0; i < 8; ++i) { a[i] = c0[i]; a[8 + i] = c1[i]; }
  return a;
}

// B fragment 32x16 (KxN), bf16. Source is N-major: srcT[n][k] contiguous in k
// (i.e. B = srcT^T). lanes 0-15: N=lane, K=0..15; lanes 16-31: N=lane-16, K=16..31.
__device__ __forceinline__ v16bf load_b(const bf16* __restrict__ srcT, int ld, int lane) {
  int n = lane & 15, half = lane >> 4;
  return *(const v16bf*)(srcT + (size_t)n * ld + 16 * half);
}

// ---------------------------------------------------------------- conversions
__global__ void cvt_f32_bf16(const float* __restrict__ in, bf16* __restrict__ out,
                             int nelem) {
  int i = blockIdx.x * blockDim.x + threadIdx.x;
  if (i < nelem) out[i] = f2bf(in[i]);
}

// out[cols_in][rows_in] = in[rows_in][cols_in]^T, fp32 -> bf16
__global__ void cvt_transpose(const float* __restrict__ in, bf16* __restrict__ out,
                              int rows_in, int cols_in) {
  int i = blockIdx.x * blockDim.x + threadIdx.x;
  if (i >= rows_in * cols_in) return;
  int c = i / rows_in, r = i % rows_in;
  out[i] = f2bf(in[(size_t)r * cols_in + c]);
}

// -------------------------------------------------- 64x64-per-wave GEMM core
// acc[i][j] += A(64xK) * Bt(64xK)^T, K = DIM. 16 WMMA per 8 fragment loads.
__device__ __forceinline__ void gemm_tile64(const bf16* __restrict__ A,
                                            const bf16* __restrict__ Bt,
                                            int lane, v8f acc[4][4]) {
  for (int kk = 0; kk < DIM; kk += 32) {
    v16bf a[4], b[4];
    #pragma unroll
    for (int i = 0; i < 4; ++i)
      a[i] = load_a(A + (size_t)(i * 16) * DIM + kk, DIM, lane);
    #pragma unroll
    for (int j = 0; j < 4; ++j)
      b[j] = load_b(Bt + (size_t)(j * 16) * DIM + kk, DIM, lane);
    // speculative prefetch of next K-chunk (global_prefetch_b8; OOB is dropped)
    __builtin_prefetch(A + (size_t)(lane & 15) * DIM + kk + 32, 0, 1);
    __builtin_prefetch(Bt + (size_t)(lane & 15) * DIM + kk + 32, 0, 1);
    #pragma unroll
    for (int i = 0; i < 4; ++i)
      #pragma unroll
      for (int j = 0; j < 4; ++j)
        acc[i][j] = wmma_bf16(a[i], b[j], acc[i][j]);
  }
}

// ------------------------------------------------------------------ QKV GEMM
// C(20480x3072) = xbf(20480x1024) * WqkvT^T; scatter to q/k (head-major) + vT.
__global__ void __launch_bounds__(128) gemm_qkv(
    const bf16* __restrict__ xbf, const bf16* __restrict__ wT,
    bf16* __restrict__ qh, bf16* __restrict__ kh, bf16* __restrict__ vT) {
  int wave = blockIdx.x * 4 + (threadIdx.x >> 5);
  int lane = threadIdx.x & 31;
  int nt = wave % (NQKV / 64);                 // 48
  int mt = wave / (NQKV / 64);
  int m0 = mt * 64, n0 = nt * 64;
  v8f acc[4][4] = {};
  gemm_tile64(xbf + (size_t)m0 * DIM, wT + (size_t)n0 * DIM, lane, acc);
  int half = lane >> 4, nl = lane & 15;
  #pragma unroll
  for (int ti = 0; ti < 4; ++ti) {
    #pragma unroll
    for (int tj = 0; tj < 4; ++tj) {
      #pragma unroll
      for (int v = 0; v < 8; ++v) {
        int row = m0 + ti * 16 + v + 8 * half;
        int col = n0 + tj * 16 + nl;
        int which = col >> 10;                 // 0:q 1:k 2:v
        int rem = col & 1023;
        int h = rem >> 6, dd = rem & 63;
        int bi = row / SEQ, s = row % SEQ;
        size_t bh = (size_t)bi * NHEADS + h;
        float val = acc[ti][tj][v];
        if (which == 0)      qh[(bh * SEQ + s) * DHEAD + dd] = f2bf(val * QSCALE);
        else if (which == 1) kh[(bh * SEQ + s) * DHEAD + dd] = f2bf(val);
        else                 vT[(bh * DHEAD + dd) * SEQ + s] = f2bf(val);
      }
    }
  }
}

// -------------------------------------------------------------- text attention
// one wave per (bh, 16-query tile): S=Q*K^T (16x256), causal+pad mask, softmax,
// O = P*V (16x64). P staged through LDS to rebuild A-fragment layout.
__global__ void __launch_bounds__(32) attn_text(
    const bf16* __restrict__ qh, const bf16* __restrict__ kh,
    const bf16* __restrict__ vT, const unsigned char* __restrict__ mask,
    bf16* __restrict__ aout) {
  __shared__ __align__(32) bf16 P[16 * TEXTLEN];
  int lane = threadIdx.x;
  int bh = blockIdx.x >> 4;
  int qt = blockIdx.x & 15;
  int bi = bh >> 4, h = bh & 15;
  int half = lane >> 4, nl = lane & 15;
  const bf16* qb = qh + ((size_t)bh * SEQ + qt * 16) * DHEAD;
  const bf16* kb = kh + (size_t)bh * SEQ * DHEAD;
  v16bf a0 = load_a(qb, DHEAD, lane);
  v16bf a1 = load_a(qb + 32, DHEAD, lane);
  v8f ac[16];
  #pragma unroll
  for (int t = 0; t < 16; ++t) {
    v8f c{};
    c = wmma_bf16(a0, load_b(kb + (size_t)t * 16 * DHEAD, DHEAD, lane), c);
    c = wmma_bf16(a1, load_b(kb + (size_t)t * 16 * DHEAD + 32, DHEAD, lane), c);
    ac[t] = c;
  }
  #pragma unroll
  for (int t = 0; t < 16; ++t) {
    int j = t * 16 + nl;
    bool alive = mask[bi * TEXTLEN + j] != 0;
    #pragma unroll
    for (int v = 0; v < 8; ++v) {
      int i = qt * 16 + v + 8 * half;
      if (j > i || !alive) ac[t][v] = NEGMAX;
    }
  }
  float inv[8];
  #pragma unroll
  for (int v = 0; v < 8; ++v) {
    float m = ac[0][v];
    #pragma unroll
    for (int t = 1; t < 16; ++t) m = fmaxf(m, ac[t][v]);
    #pragma unroll
    for (int off = 1; off < 16; off <<= 1) m = fmaxf(m, __shfl_xor(m, off, 32));
    float s = 0.f;
    #pragma unroll
    for (int t = 0; t < 16; ++t) { float e = __expf(ac[t][v] - m); ac[t][v] = e; s += e; }
    #pragma unroll
    for (int off = 1; off < 16; off <<= 1) s += __shfl_xor(s, off, 32);
    inv[v] = 1.f / s;
  }
  #pragma unroll
  for (int t = 0; t < 16; ++t)
    #pragma unroll
    for (int v = 0; v < 8; ++v)
      P[(v + 8 * half) * TEXTLEN + t * 16 + nl] = f2bf(ac[t][v] * inv[v]);
  __syncthreads();
  const bf16* vb = vT + (size_t)bh * DHEAD * SEQ;
  v8f oc[4] = {};
  #pragma unroll
  for (int c8 = 0; c8 < 8; ++c8) {
    v16bf ap = load_a(P + c8 * 32, TEXTLEN, lane);
    #pragma unroll
    for (int t = 0; t < 4; ++t)
      oc[t] = wmma_bf16(ap, load_b(vb + (size_t)t * 16 * SEQ + c8 * 32, SEQ, lane), oc[t]);
  }
  #pragma unroll
  for (int t = 0; t < 4; ++t)
    #pragma unroll
    for (int v = 0; v < 8; ++v) {
      int s = qt * 16 + v + 8 * half;
      int col = h * DHEAD + t * 16 + nl;
      aout[((size_t)bi * SEQ + s) * DIM + col] = f2bf(oc[t][v]);
    }
}

// ------------------------------------------------------- axial image attention
// one wave per (bh, image row r, 16-query half): keys = 256 text + 32 in-row
// causal image keys -> 18 S tiles.
__global__ void __launch_bounds__(32) attn_img(
    const bf16* __restrict__ qh, const bf16* __restrict__ kh,
    const bf16* __restrict__ vT, const unsigned char* __restrict__ mask,
    bf16* __restrict__ aout) {
  __shared__ __align__(32) bf16 P[16 * 288];
  int lane = threadIdx.x;
  int id = blockIdx.x;
  int mtile = id & 1;
  int r = (id >> 1) & 31;
  int bh = id >> 6;
  int bi = bh >> 4, h = bh & 15;
  int half = lane >> 4, nl = lane & 15;
  int qseq = TEXTLEN + r * IMGSIDE + mtile * 16;
  const bf16* qb = qh + ((size_t)bh * SEQ + qseq) * DHEAD;
  const bf16* kb = kh + (size_t)bh * SEQ * DHEAD;
  v16bf a0 = load_a(qb, DHEAD, lane);
  v16bf a1 = load_a(qb + 32, DHEAD, lane);
  v8f ac[18];
  #pragma unroll
  for (int t = 0; t < 18; ++t) {
    int jb = (t < 16) ? t * 16 : TEXTLEN + r * IMGSIDE + (t - 16) * 16;
    v8f c{};
    c = wmma_bf16(a0, load_b(kb + (size_t)jb * DHEAD, DHEAD, lane), c);
    c = wmma_bf16(a1, load_b(kb + (size_t)jb * DHEAD + 32, DHEAD, lane), c);
    ac[t] = c;
  }
  #pragma unroll
  for (int t = 0; t < 18; ++t) {
    if (t < 16) {
      int j = t * 16 + nl;
      if (!mask[bi * TEXTLEN + j]) {
        #pragma unroll
        for (int v = 0; v < 8; ++v) ac[t][v] = NEGMAX;
      }
    } else {
      int jj = (t - 16) * 16 + nl;
      #pragma unroll
      for (int v = 0; v < 8; ++v) {
        int i = mtile * 16 + v + 8 * half;
        if (jj > i) ac[t][v] = NEGMAX;
      }
    }
  }
  float inv[8];
  #pragma unroll
  for (int v = 0; v < 8; ++v) {
    float m = ac[0][v];
    #pragma unroll
    for (int t = 1; t < 18; ++t) m = fmaxf(m, ac[t][v]);
    #pragma unroll
    for (int off = 1; off < 16; off <<= 1) m = fmaxf(m, __shfl_xor(m, off, 32));
    float s = 0.f;
    #pragma unroll
    for (int t = 0; t < 18; ++t) { float e = __expf(ac[t][v] - m); ac[t][v] = e; s += e; }
    #pragma unroll
    for (int off = 1; off < 16; off <<= 1) s += __shfl_xor(s, off, 32);
    inv[v] = 1.f / s;
  }
  #pragma unroll
  for (int t = 0; t < 18; ++t)
    #pragma unroll
    for (int v = 0; v < 8; ++v)
      P[(v + 8 * half) * 288 + t * 16 + nl] = f2bf(ac[t][v] * inv[v]);
  __syncthreads();
  const bf16* vb = vT + (size_t)bh * DHEAD * SEQ;
  v8f oc[4] = {};
  #pragma unroll
  for (int c9 = 0; c9 < 9; ++c9) {
    int jb = (c9 < 8) ? c9 * 32 : TEXTLEN + r * IMGSIDE;
    v16bf ap = load_a(P + c9 * 32, 288, lane);
    #pragma unroll
    for (int t = 0; t < 4; ++t)
      oc[t] = wmma_bf16(ap, load_b(vb + (size_t)t * 16 * SEQ + jb, SEQ, lane), oc[t]);
  }
  #pragma unroll
  for (int t = 0; t < 4; ++t)
    #pragma unroll
    for (int v = 0; v < 8; ++v) {
      int s = qseq + v + 8 * half;
      int col = h * DHEAD + t * 16 + nl;
      aout[((size_t)bi * SEQ + s) * DIM + col] = f2bf(oc[t][v]);
    }
}

// --------------------------------------------------------- output projection
__global__ void __launch_bounds__(128) gemm_out(
    const bf16* __restrict__ ain, const bf16* __restrict__ wT,
    const float* __restrict__ bout, float* __restrict__ out) {
  int wave = blockIdx.x * 4 + (threadIdx.x >> 5);
  int lane = threadIdx.x & 31;
  int nt = wave % (DIM / 64);                  // 16
  int mt = wave / (DIM / 64);
  int m0 = mt * 64, n0 = nt * 64;
  v8f acc[4][4] = {};
  gemm_tile64(ain + (size_t)m0 * DIM, wT + (size_t)n0 * DIM, lane, acc);
  int half = lane >> 4, nl = lane & 15;
  #pragma unroll
  for (int ti = 0; ti < 4; ++ti) {
    #pragma unroll
    for (int tj = 0; tj < 4; ++tj) {
      int col = n0 + tj * 16 + nl;
      float bias = bout[col];
      #pragma unroll
      for (int v = 0; v < 8; ++v) {
        int row = m0 + ti * 16 + v + 8 * half;
        out[(size_t)row * DIM + col] = acc[ti][tj][v] + bias;
      }
    }
  }
}

extern "C" void kernel_launch(void* const* d_in, const int* in_sizes, int n_in,
                              void* d_out, int out_size, void* d_ws, size_t ws_size,
                              hipStream_t stream) {
  (void)in_sizes; (void)n_in; (void)out_size; (void)ws_size;
  const float* x            = (const float*)d_in[0];
  const unsigned char* mask = (const unsigned char*)d_in[1];
  const float* Wqkv         = (const float*)d_in[2];
  const float* Wout         = (const float*)d_in[3];
  const float* bout         = (const float*)d_in[4];
  float* out = (float*)d_out;

  char* ws = (char*)d_ws;
  size_t off = 0;
  auto take = [&](size_t bytes) {
    char* p = ws + off;
    off = (off + bytes + 255) & ~(size_t)255;
    return p;
  };
  bf16* xbf  = (bf16*)take((size_t)MROWS * DIM * 2);
  bf16* wqT  = (bf16*)take((size_t)NQKV * DIM * 2);
  bf16* woT  = (bf16*)take((size_t)DIM * DIM * 2);
  bf16* qh   = (bf16*)take((size_t)BH * SEQ * DHEAD * 2);
  bf16* kh   = (bf16*)take((size_t)BH * SEQ * DHEAD * 2);
  bf16* vT   = (bf16*)take((size_t)BH * DHEAD * SEQ * 2);
  bf16* aout = (bf16*)take((size_t)MROWS * DIM * 2);

  cvt_f32_bf16<<<(MROWS * DIM + 255) / 256, 256, 0, stream>>>(x, xbf, MROWS * DIM);
  cvt_transpose<<<(NQKV * DIM + 255) / 256, 256, 0, stream>>>(Wqkv, wqT, DIM, NQKV);
  cvt_transpose<<<(DIM * DIM + 255) / 256, 256, 0, stream>>>(Wout, woT, DIM, DIM);

  gemm_qkv<<<(MROWS / 64) * (NQKV / 64) / 4, 128, 0, stream>>>(xbf, wqT, qh, kh, vT);
  attn_text<<<BH * 16, 32, 0, stream>>>(qh, kh, vT, mask, aout);
  attn_img<<<BH * IMGSIDE * 2, 32, 0, stream>>>(qh, kh, vT, mask, aout);
  gemm_out<<<(MROWS / 64) * (DIM / 64) / 4, 128, 0, stream>>>(aout, woT, bout, out);
}
